// GraphTopoLayer_22110491640201
// MI455X (gfx1250) — matrix operand order repaired and checked by reading, not compiled, not run on hardware
//
#include <hip/hip_runtime.h>

#define HIDDEN   128
#define EDGE_DIM 259
#define EPAD     288   // edge dim padded to 9 chunks of K=32
#define A1PAD    272   // Wa1 out-dim padded to 17 tiles of 16
#define WAVES    4

typedef __attribute__((ext_vector_type(16))) __bf16 v16bf;
typedef __attribute__((ext_vector_type(8)))  float  v8f;
typedef unsigned int __attribute__((ext_vector_type(4), may_alias)) u32x4;

static __device__ __forceinline__ unsigned short f2bf(float f) {
  unsigned int u = __float_as_uint(f);
  u += 0x7FFFu + ((u >> 16) & 1u);          // round-to-nearest-even
  return (unsigned short)(u >> 16);
}

// Build a v16bf operand from two 16-byte contiguous runs in LDS.
static __device__ __forceinline__ v16bf ld_frag(const unsigned short* p0,
                                                const unsigned short* p1) {
  union { u32x4 u[2]; v16bf v; } t;
  t.u[0] = *(const u32x4*)p0;
  t.u[1] = *(const u32x4*)p1;
  return t.v;
}

// ---------------- weight conversion: f32 -> padded bf16 (+ padded f32 biases) -------
__global__ void prep_weights(const float* __restrict__ Wa1, const float* __restrict__ Wm1,
                             const float* __restrict__ Wm2, const float* __restrict__ ba1,
                             const float* __restrict__ Wa2, unsigned short* __restrict__ wsW) {
  unsigned short* wa1 = wsW;                       // [A1PAD][EPAD] bf16 (n-major)
  unsigned short* wm1 = wa1 + EPAD * A1PAD;        // [HIDDEN][EPAD] bf16 (n-major)
  unsigned short* wm2 = wm1 + EPAD * HIDDEN;       // [HIDDEN][HIDDEN] bf16
  float* ba1p = (float*)(wm2 + HIDDEN * HIDDEN);   // [A1PAD] f32, zero-padded
  float* wa2p = ba1p + A1PAD;                      // [A1PAD] f32, zero-padded
  const int stride = gridDim.x * blockDim.x;
  const int t0 = blockIdx.x * blockDim.x + threadIdx.x;
  for (int t = t0; t < EPAD * A1PAD; t += stride) {
    int r = t / EPAD, c = t - r * EPAD;
    wa1[t] = (r < EDGE_DIM && c < EDGE_DIM) ? f2bf(Wa1[c * EDGE_DIM + r]) : (unsigned short)0;
  }
  for (int t = t0; t < EPAD * HIDDEN; t += stride) {
    int r = t / EPAD, c = t - r * EPAD;
    wm1[t] = (c < EDGE_DIM) ? f2bf(Wm1[c * HIDDEN + r]) : (unsigned short)0;
  }
  for (int t = t0; t < HIDDEN * HIDDEN; t += stride) wm2[t] = f2bf(Wm2[(t & 127) * HIDDEN + (t >> 7)]);
  for (int t = t0; t < A1PAD; t += stride) {
    ba1p[t] = (t < EDGE_DIM) ? ba1[t] : 0.f;
    wa2p[t] = (t < EDGE_DIM) ? Wa2[t] : 0.f;
  }
}

// ---------------- exact top-16 kNN (set only; order is irrelevant downstream) -------
__global__ void __launch_bounds__(256) knn16(const float* __restrict__ xyz,
                                             int* __restrict__ idx, int N) {
  const int b = blockIdx.y;
  const int n = blockIdx.x * 256 + threadIdx.x;
  const float* X = xyz + (size_t)b * N * 3;
  const float qx = X[n * 3 + 0], qy = X[n * 3 + 1], qz = X[n * 3 + 2];
  __shared__ float sx[768];
  float bd[16]; int bi[16];
#pragma unroll
  for (int s = 0; s < 16; ++s) { bd[s] = 3.4e38f; bi[s] = 0; }
  float wd = 3.4e38f;
  for (int t0 = 0; t0 < N; t0 += 256) {
    __syncthreads();
    const int c = t0 + threadIdx.x;
    sx[threadIdx.x]       = X[c * 3 + 0];
    sx[256 + threadIdx.x] = X[c * 3 + 1];
    sx[512 + threadIdx.x] = X[c * 3 + 2];
    __syncthreads();
    for (int j = 0; j < 256; ++j) {
      const float dx = sx[j] - qx, dy = sx[256 + j] - qy, dz = sx[512 + j] - qz;
      const float d2 = dx * dx + dy * dy + dz * dz;
      if (d2 < wd) {
        bool done = false;
#pragma unroll
        for (int s = 0; s < 16; ++s)
          if (!done && bd[s] == wd) { bd[s] = d2; bi[s] = t0 + j; done = true; }
        wd = bd[0];
#pragma unroll
        for (int s = 1; s < 16; ++s) wd = fmaxf(wd, bd[s]);
      }
    }
  }
  int* op = idx + ((size_t)b * N + n) * 16;
#pragma unroll
  for (int s = 0; s < 16; ++s) op[s] = bi[s];
}

// ---------------- fused edge-MLP + attention + message + aggregate ------------------
__global__ void __launch_bounds__(WAVES * 32)
fused_gnn(const float* __restrict__ xyz, const float* __restrict__ h,
          const float* __restrict__ bm1, const float* __restrict__ bm2,
          const int* __restrict__ idx, const unsigned short* __restrict__ wsW,
          const float* __restrict__ ba1p, const float* __restrict__ wa2p,
          float* __restrict__ out, int B, int N) {
  extern __shared__ unsigned short sm[];
  unsigned short* sWa1 = sm;                              // [A1PAD][EPAD]
  unsigned short* sWm1 = sWa1 + EPAD * A1PAD;             // [HIDDEN][EPAD]
  unsigned short* sWm2 = sWm1 + EPAD * HIDDEN;            // [HIDDEN][HIDDEN]
  unsigned short* sEb  = sWm2 + HIDDEN * HIDDEN;          // WAVES * [16][EPAD]
  unsigned short* sM1b = sEb + WAVES * 16 * EPAD;         // WAVES * [16][HIDDEN]

  const int tid = threadIdx.x;
  const int wave = tid >> 5, lane = tid & 31;
  const int lo = lane & 15, hi = lane >> 4;               // C-layout: M = r + 8*hi, N = lo

  { // cooperative weight load (bf16, already padded & laid out in ws)
    const int n16 = (EPAD * A1PAD + EPAD * HIDDEN + HIDDEN * HIDDEN) / 8;  // u32x4 units
    const u32x4* src = (const u32x4*)wsW;
    u32x4* dst = (u32x4*)sm;
    for (int t = tid; t < n16; t += blockDim.x) dst[t] = src[t];
  }
  __syncthreads();

  unsigned short* sE  = sEb  + wave * 16 * EPAD;
  unsigned short* sM1 = sM1b + wave * 16 * HIDDEN;
  const int BN = B * N;

  for (int node = blockIdx.x * WAVES + wave; node < BN; node += gridDim.x * WAVES) {
    const int b = node / N;
    const float* hc = h + (size_t)node * HIDDEN;

    float ch[4];
#pragma unroll
    for (int j = 0; j < 4; ++j) ch[j] = hc[lane + 32 * j];

    // ---- build E = [h_c | h_n - h_c | rel_xyz | 0...] (16 x EPAD bf16) ----
    for (int m = 0; m < 16; ++m) {
      const int nb = idx[(size_t)node * 16 + m];
      const float* hn = h + ((size_t)b * N + nb) * HIDDEN;
      unsigned short* er = sE + m * EPAD;
#pragma unroll
      for (int j = 0; j < 4; ++j) {
        float nv = hn[lane + 32 * j];
        er[lane + 32 * j] = f2bf(ch[j]);
        er[HIDDEN + lane + 32 * j] = f2bf(nv - ch[j]);
      }
      er[256 + lane] = (lane < 3)
          ? f2bf(xyz[((size_t)b * N + nb) * 3 + lane] - xyz[(size_t)node * 3 + lane])
          : (unsigned short)0;
    }
    asm volatile("s_wait_dscnt 0" ::: "memory");

    // ---- preload the 9 A-fragments of E (lane = row M=lo; two 8-half runs) ----
    v16bf af[9];
#pragma unroll
    for (int kc = 0; kc < 9; ++kc) {
      const unsigned short* p = sE + lo * EPAD + kc * 32 + hi * 8;
      af[kc] = ld_frag(p, p + 16);
    }

    // ---- attention scores: relu(E*Wa1 + ba1) . Wa2  (ba2 cancels in softmax) ----
    float ps[8];
#pragma unroll
    for (int r = 0; r < 8; ++r) ps[r] = 0.f;
#pragma unroll 1
    for (int nt = 0; nt < 17; ++nt) {
      const unsigned short* pb = sWa1 + (nt * 16 + lo) * EPAD + hi * 8;  // chunk kc at +kc*32
      v8f c = {};
      v16bf bw = ld_frag(pb, pb + 16);                // prefetch kc=0
#pragma unroll
      for (int kc = 0; kc < 9; ++kc) {
        v16bf bn = bw;
        if (kc < 8) {                                  // prefetch kc+1 before consuming kc
          const unsigned short* q = pb + (kc + 1) * 32;
          bn = ld_frag(q, q + 16);
        }
        c = __builtin_amdgcn_wmma_f32_16x16x32_bf16(false, af[kc], false, bw,
                                                    (short)0, c, false, false);
        bw = bn;
      }
      const int col = nt * 16 + lo;
      const float bias = ba1p[col];                    // zero-padded: no branches
      const float w2   = wa2p[col];
#pragma unroll
      for (int r = 0; r < 8; ++r) ps[r] = fmaf(fmaxf(c[r] + bias, 0.f), w2, ps[r]);
    }
#pragma unroll
    for (int r = 0; r < 8; ++r) {
      float v = ps[r];
      v += __shfl_xor(v, 1, 32); v += __shfl_xor(v, 2, 32);
      v += __shfl_xor(v, 4, 32); v += __shfl_xor(v, 8, 32);
      ps[r] = v;                                   // full score for M = r + 8*hi
    }
    float os[8];
#pragma unroll
    for (int r = 0; r < 8; ++r) os[r] = __shfl_xor(ps[r], 16, 32);  // other half's rows
    float mx = ps[0];
#pragma unroll
    for (int r = 1; r < 8; ++r) mx = fmaxf(mx, ps[r]);
#pragma unroll
    for (int r = 0; r < 8; ++r) mx = fmaxf(mx, os[r]);
    float den = 0.f, al[8];
#pragma unroll
    for (int r = 0; r < 8; ++r) { al[r] = __expf(ps[r] - mx); den += al[r]; }
#pragma unroll
    for (int r = 0; r < 8; ++r) den += __expf(os[r] - mx);
    const float rden = 1.f / den;
#pragma unroll
    for (int r = 0; r < 8; ++r) al[r] *= rden;     // alpha for rows M = r + 8*hi

    // ---- M1 = relu(E*Wm1 + bm1) -> LDS bf16 ----
#pragma unroll 1
    for (int nt = 0; nt < 8; ++nt) {
      const unsigned short* pb = sWm1 + (nt * 16 + lo) * EPAD + hi * 8;
      v8f c = {};
      v16bf bw = ld_frag(pb, pb + 16);
#pragma unroll
      for (int kc = 0; kc < 9; ++kc) {
        v16bf bn = bw;
        if (kc < 8) {
          const unsigned short* q = pb + (kc + 1) * 32;
          bn = ld_frag(q, q + 16);
        }
        c = __builtin_amdgcn_wmma_f32_16x16x32_bf16(false, af[kc], false, bw,
                                                    (short)0, c, false, false);
        bw = bn;
      }
      const int col = nt * 16 + lo;
      const float bias = bm1[col];
#pragma unroll
      for (int r = 0; r < 8; ++r)
        sM1[(r + 8 * hi) * HIDDEN + col] = f2bf(fmaxf(c[r] + bias, 0.f));
    }
    asm volatile("s_wait_dscnt 0" ::: "memory");

    v16bf mf[4];
#pragma unroll
    for (int kc = 0; kc < 4; ++kc) {
      const unsigned short* p = sM1 + lo * HIDDEN + kc * 32 + hi * 8;
      mf[kc] = ld_frag(p, p + 16);
    }

    // ---- out = h + bm2 + alpha^T (M1*Wm2)   (sum(alpha)=1 absorbs bm2) ----
    float* outp = out + (size_t)node * HIDDEN;
#pragma unroll 1
    for (int nt = 0; nt < 8; ++nt) {
      const unsigned short* pb = sWm2 + (nt * 16 + lo) * HIDDEN + hi * 8;
      v8f c = {};
      v16bf bw = ld_frag(pb, pb + 16);
#pragma unroll
      for (int kc = 0; kc < 4; ++kc) {
        v16bf bn = bw;
        if (kc < 3) {
          const unsigned short* q = pb + (kc + 1) * 32;
          bn = ld_frag(q, q + 16);
        }
        c = __builtin_amdgcn_wmma_f32_16x16x32_bf16(false, mf[kc], false, bw,
                                                    (short)0, c, false, false);
        bw = bn;
      }
      float pacc = 0.f;
#pragma unroll
      for (int r = 0; r < 8; ++r) pacc = fmaf(al[r], c[r], pacc);
      pacc += __shfl_xor(pacc, 16, 32);            // combine M 0-7 with M 8-15
      const int col = nt * 16 + lo;
      if (hi == 0) outp[col] = hc[col] + bm2[col] + pacc;
    }
  }
}

extern "C" void kernel_launch(void* const* d_in, const int* in_sizes, int n_in,
                              void* d_out, int out_size, void* d_ws, size_t ws_size,
                              hipStream_t stream) {
  const float* xyz = (const float*)d_in[0];
  const float* h   = (const float*)d_in[1];
  const float* Wa1 = (const float*)d_in[2];
  const float* ba1 = (const float*)d_in[3];
  const float* Wa2 = (const float*)d_in[4];
  // d_in[5] = ba2: constant shift of all scores, cancels in softmax
  const float* Wm1 = (const float*)d_in[6];
  const float* bm1 = (const float*)d_in[7];
  const float* Wm2 = (const float*)d_in[8];
  const float* bm2 = (const float*)d_in[9];
  float* out = (float*)d_out;

  const int B = 8;
  const int BN = in_sizes[0] / 3;   // B*N
  const int N = BN / B;             // 4096

  // workspace: [idx int32 BN*16][wa1bf|wm1bf|wm2bf bf16][ba1p|wa2p f32] (~2.4 MB)
  int* idx = (int*)d_ws;
  unsigned short* wsW = (unsigned short*)((char*)d_ws + (size_t)BN * 16 * sizeof(int));
  const float* ba1p = (const float*)(wsW + EPAD * A1PAD + EPAD * HIDDEN + HIDDEN * HIDDEN);
  const float* wa2p = ba1p + A1PAD;

  prep_weights<<<dim3(256), dim3(256), 0, stream>>>(Wa1, Wm1, Wm2, ba1, Wa2, wsW);
  knn16<<<dim3(N / 256, B), dim3(256), 0, stream>>>(xyz, idx, N);

  const int smem = (EPAD * A1PAD + EPAD * HIDDEN + HIDDEN * HIDDEN +
                    WAVES * 16 * EPAD + WAVES * 16 * HIDDEN) * 2;  // 316,416 B < 320 KB
  fused_gnn<<<dim3(512), dim3(WAVES * 32), smem, stream>>>(xyz, h, bm1, bm2, idx, wsW,
                                                           ba1p, wa2p, out, B, N);
}
